// TransformerBlock_88424786690745
// MI455X (gfx1250) — matrix operand order repaired and checked by reading, not compiled, not run on hardware
//
#include <hip/hip_runtime.h>

typedef __attribute__((ext_vector_type(16))) __bf16 v16bf;
typedef __attribute__((ext_vector_type(8)))  __bf16 v8bf;
typedef __attribute__((ext_vector_type(8)))  float  v8f;

#define LDST 48   // bf16 LDS row stride for GEMM tiles (96B rows, 16B aligned)
#define FST  72   // bf16 LDS row stride for attention tiles (144B rows, 16B aligned)

__device__ __forceinline__ unsigned short f2bf(float f) {
  unsigned u = __float_as_uint(f);
  u += 0x7FFFu + ((u >> 16) & 1u);           // round-to-nearest-even
  return (unsigned short)(u >> 16);
}
__device__ __forceinline__ unsigned pack2(unsigned short a, unsigned short b) {
  return (unsigned)a | ((unsigned)b << 16);
}
__device__ __forceinline__ v16bf cat8(v8bf lo, v8bf hi) {
  return __builtin_shufflevector(lo, hi, 0,1,2,3,4,5,6,7,8,9,10,11,12,13,14,15);
}

// ---------------------------------------------------------------------------
// RMSNorm: one block per row of D floats
// ---------------------------------------------------------------------------
__global__ __launch_bounds__(256) void rmsnorm_kernel(
    const float* __restrict__ x, const float* __restrict__ g,
    float* __restrict__ out, int D) {
  __shared__ float red[8];
  const int row = blockIdx.x;
  const float* xr = x + (size_t)row * D;
  float ss = 0.f;
  for (int i = threadIdx.x; i < D; i += 256) { float v = xr[i]; ss += v * v; }
#pragma unroll
  for (int d = 16; d >= 1; d >>= 1) ss += __shfl_xor(ss, d, 32);
  if ((threadIdx.x & 31) == 0) red[threadIdx.x >> 5] = ss;
  __syncthreads();
  float tot = 0.f;
#pragma unroll
  for (int w = 0; w < 8; ++w) tot += red[w];
  const float inv = rsqrtf(tot / (float)D + 1e-6f);
  for (int i = threadIdx.x; i < D; i += 256)
    out[(size_t)row * D + i] = g[i] * xr[i] * inv;
}

// ---------------------------------------------------------------------------
// Tiled WMMA GEMM: C[M,N] = A[M,K] * B[K,N], f32 in/out, bf16 matrix core.
// Block = 256 threads (8 waves, 2x4), tile 128x128, K-step 32.
// MODE 0: C = acc ; MODE 1: C = acc + X ; MODE 2: C += X[m*rs]*acc
// ---------------------------------------------------------------------------
template <int MODE>
__global__ __launch_bounds__(256) void gemm_bf16_wmma(
    const float* __restrict__ A, const float* __restrict__ B,
    float* __restrict__ C, const float* __restrict__ X,
    int M, int N, int K, int rs) {
  __shared__ unsigned short ldsA[128 * LDST];   // [m][k]
  __shared__ unsigned short ldsB[128 * LDST];   // transposed: [n][k]
  const int t = threadIdx.x, lane = t & 31, wave = t >> 5;
  const int wm = wave & 1, wn = wave >> 1;      // 2 waves (M) x 4 waves (N)
  const int hl = lane >> 4, l16 = lane & 15;
  const int bm = blockIdx.y * 128, bn = blockIdx.x * 128;

  v8f acc[4][2];
#pragma unroll
  for (int i = 0; i < 4; ++i)
#pragma unroll
    for (int j = 0; j < 2; ++j)
#pragma unroll
      for (int r = 0; r < 8; ++r) acc[i][j][r] = 0.f;

  for (int k0 = 0; k0 < K; k0 += 32) {
    // ---- stage A tile 128x32 (f32 -> bf16) ----
#pragma unroll
    for (int i = 0; i < 4; ++i) {
      const int f = t + 256 * i;
      const int r = f >> 3, c4 = (f & 7) * 4;
      const float4 va = *reinterpret_cast<const float4*>(
          A + (size_t)(bm + r) * K + k0 + c4);
      *reinterpret_cast<unsigned*>(&ldsA[r * LDST + c4])     = pack2(f2bf(va.x), f2bf(va.y));
      *reinterpret_cast<unsigned*>(&ldsA[r * LDST + c4 + 2]) = pack2(f2bf(va.z), f2bf(va.w));
      if (k0 + 32 < K)
        __builtin_prefetch(A + (size_t)(bm + r) * K + k0 + 32 + c4, 0, 1);
    }
    // ---- stage B tile 32x128 transposed into [n][k] ----
#pragma unroll
    for (int i = 0; i < 4; ++i) {
      const int f = t + 256 * i;
      const int kk = f >> 5, c4 = (f & 31) * 4;
      const float4 vb = *reinterpret_cast<const float4*>(
          B + (size_t)(k0 + kk) * N + bn + c4);
      ldsB[(c4 + 0) * LDST + kk] = f2bf(vb.x);
      ldsB[(c4 + 1) * LDST + kk] = f2bf(vb.y);
      ldsB[(c4 + 2) * LDST + kk] = f2bf(vb.z);
      ldsB[(c4 + 3) * LDST + kk] = f2bf(vb.w);
      if (k0 + 32 < K)
        __builtin_prefetch(B + (size_t)(k0 + 32 + kk) * N + bn + c4, 0, 1);
    }
    __syncthreads();

    // ---- fragments (A: 16x32 layout, B: 32x16 layout per ISA 7.12.2) ----
    const int kb = hl * 8;
    v16bf afrag[4];
#pragma unroll
    for (int i = 0; i < 4; ++i) {
      const int m = wm * 64 + i * 16 + l16;
      v8bf lo = *reinterpret_cast<const v8bf*>(&ldsA[m * LDST + kb]);
      v8bf hi = *reinterpret_cast<const v8bf*>(&ldsA[m * LDST + kb + 16]);
      afrag[i] = cat8(lo, hi);
    }
    v16bf bfrag[2];
#pragma unroll
    for (int j = 0; j < 2; ++j) {
      const int n = wn * 32 + j * 16 + l16;
      bfrag[j] = cat8(*reinterpret_cast<const v8bf*>(&ldsB[n * LDST + kb]),
                      *reinterpret_cast<const v8bf*>(&ldsB[n * LDST + kb + 16]));
    }
#pragma unroll
    for (int i = 0; i < 4; ++i)
#pragma unroll
      for (int j = 0; j < 2; ++j)
        acc[i][j] = __builtin_amdgcn_wmma_f32_16x16x32_bf16(
            false, afrag[i], false, bfrag[j], (short)0, acc[i][j], false, false);
    __syncthreads();
  }

  // ---- epilogue ----
#pragma unroll
  for (int i = 0; i < 4; ++i)
#pragma unroll
    for (int j = 0; j < 2; ++j)
#pragma unroll
      for (int r = 0; r < 8; ++r) {
        const int m = bm + wm * 64 + i * 16 + r + 8 * hl;
        const int n = bn + wn * 32 + j * 16 + l16;
        const size_t idx = (size_t)m * N + n;
        float v = acc[i][j][r];
        if (MODE == 1) v += X[idx];
        if (MODE == 2) v = C[idx] + X[(size_t)m * rs] * v;
        C[idx] = v;
      }
}

// ---------------------------------------------------------------------------
// Flash attention: grid (S/64, B*NH), 128 threads (4 waves, 16 q-rows each).
// HD = 64, online softmax, WMMA for QK^T and PV.
// ---------------------------------------------------------------------------
__global__ __launch_bounds__(128) void flash_attn_kernel(
    const float* __restrict__ Q, const float* __restrict__ Kg,
    const float* __restrict__ Vg, const int* __restrict__ mask,
    float* __restrict__ O) {
  constexpr int S = 1024, D = 1024, NH = 16, HD = 64;
  __shared__ unsigned short ldsQ[64 * FST];   // [q][d]
  __shared__ unsigned short ldsK[64 * FST];   // [n][d]  (== B^T for scores)
  __shared__ unsigned short ldsVt[64 * FST];  // [d][n]  (== B^T for PV)
  __shared__ unsigned short ldsP[64 * FST];   // [q][n]
  const int t = threadIdx.x, lane = t & 31, wave = t >> 5;
  const int hl = lane >> 4, l16 = lane & 15;
  const int bh = blockIdx.y, bb = bh >> 4, hh = bh & (NH - 1);
  const int q0 = blockIdx.x * 64;
  const size_t baseQ = ((size_t)bb * S + q0) * D + hh * HD;

  // stage Q tile 64x64
#pragma unroll
  for (int i = 0; i < 8; ++i) {
    const int f = t + 128 * i;
    const int r = f >> 4, c4 = (f & 15) * 4;
    const float4 v = *reinterpret_cast<const float4*>(Q + baseQ + (size_t)r * D + c4);
    *reinterpret_cast<unsigned*>(&ldsQ[r * FST + c4])     = pack2(f2bf(v.x), f2bf(v.y));
    *reinterpret_cast<unsigned*>(&ldsQ[r * FST + c4 + 2]) = pack2(f2bf(v.z), f2bf(v.w));
  }
  __syncthreads();
  v16bf qa[2];
  {
    const int m = wave * 16 + l16;
#pragma unroll
    for (int s = 0; s < 2; ++s) {
      v8bf lo = *reinterpret_cast<const v8bf*>(&ldsQ[m * FST + s * 32 + hl * 8]);
      v8bf hi = *reinterpret_cast<const v8bf*>(&ldsQ[m * FST + s * 32 + hl * 8 + 16]);
      qa[s] = cat8(lo, hi);
    }
  }

  float mrow[8], lrow[8];
  v8f oacc[4];
#pragma unroll
  for (int r = 0; r < 8; ++r) { mrow[r] = -1e30f; lrow[r] = 0.f; }
#pragma unroll
  for (int j = 0; j < 4; ++j)
#pragma unroll
    for (int r = 0; r < 8; ++r) oacc[j][r] = 0.f;

  for (int c = 0; c < S / 64; ++c) {
    const int k0c = c * 64;
    __syncthreads();  // previous iter's LDS consumers done
    const size_t baseK = ((size_t)bb * S + k0c) * D + hh * HD;
#pragma unroll
    for (int i = 0; i < 8; ++i) {
      const int f = t + 128 * i;
      const int r = f >> 4, c4 = (f & 15) * 4;
      const float4 kv = *reinterpret_cast<const float4*>(Kg + baseK + (size_t)r * D + c4);
      *reinterpret_cast<unsigned*>(&ldsK[r * FST + c4])     = pack2(f2bf(kv.x), f2bf(kv.y));
      *reinterpret_cast<unsigned*>(&ldsK[r * FST + c4 + 2]) = pack2(f2bf(kv.z), f2bf(kv.w));
      const float4 vv = *reinterpret_cast<const float4*>(Vg + baseK + (size_t)r * D + c4);
      ldsVt[(c4 + 0) * FST + r] = f2bf(vv.x);
      ldsVt[(c4 + 1) * FST + r] = f2bf(vv.y);
      ldsVt[(c4 + 2) * FST + r] = f2bf(vv.z);
      ldsVt[(c4 + 3) * FST + r] = f2bf(vv.w);
    }
    __syncthreads();

    // scores S = Q K^T * scale  (4 n-tiles of 16, K-dim 64 = 2 wmma steps)
    v8f st[4];
#pragma unroll
    for (int j = 0; j < 4; ++j) {
#pragma unroll
      for (int r = 0; r < 8; ++r) st[j][r] = 0.f;
#pragma unroll
      for (int s = 0; s < 2; ++s) {
        v16bf kf = cat8(*reinterpret_cast<const v8bf*>(
            &ldsK[(j * 16 + l16) * FST + s * 32 + hl * 8]),
            *reinterpret_cast<const v8bf*>(
            &ldsK[(j * 16 + l16) * FST + s * 32 + hl * 8 + 16]));
        st[j] = __builtin_amdgcn_wmma_f32_16x16x32_bf16(
            false, qa[s], false, kf, (short)0, st[j], false, false);
      }
    }
    // scale + mask
    const int qrb = q0 + wave * 16 + 8 * hl;
#pragma unroll
    for (int j = 0; j < 4; ++j) {
      const int kk = k0c + j * 16 + l16;
#pragma unroll
      for (int r = 0; r < 8; ++r) {
        float sv = st[j][r] * 0.125f;  // HD^-0.5
        if (mask[(size_t)(qrb + r) * S + kk] == 0) sv = -1e9f;
        st[j][r] = sv;
      }
    }
    // online softmax per row
#pragma unroll
    for (int r = 0; r < 8; ++r) {
      float rm = fmaxf(fmaxf(st[0][r], st[1][r]), fmaxf(st[2][r], st[3][r]));
#pragma unroll
      for (int d = 8; d >= 1; d >>= 1) rm = fmaxf(rm, __shfl_xor(rm, d, 32));
      const float mnew = fmaxf(mrow[r], rm);
      const float alpha = __expf(mrow[r] - mnew);
      float rsum = 0.f;
#pragma unroll
      for (int j = 0; j < 4; ++j) {
        const float p = __expf(st[j][r] - mnew);
        st[j][r] = p; rsum += p;
      }
#pragma unroll
      for (int d = 8; d >= 1; d >>= 1) rsum += __shfl_xor(rsum, d, 32);
      lrow[r] = lrow[r] * alpha + rsum;
      mrow[r] = mnew;
#pragma unroll
      for (int j = 0; j < 4; ++j) oacc[j][r] *= alpha;
    }
    // write P (bf16) to LDS to re-shape into A-fragment layout
#pragma unroll
    for (int j = 0; j < 4; ++j)
#pragma unroll
      for (int r = 0; r < 8; ++r)
        ldsP[(wave * 16 + r + 8 * hl) * FST + j * 16 + l16] = f2bf(st[j][r]);
    __syncthreads();
    // O += P V  (K-dim 64 = 2 wmma steps, 4 d-tiles)
#pragma unroll
    for (int s = 0; s < 2; ++s) {
      const int m = wave * 16 + l16;
      v8bf lo = *reinterpret_cast<const v8bf*>(&ldsP[m * FST + s * 32 + hl * 8]);
      v8bf hi = *reinterpret_cast<const v8bf*>(&ldsP[m * FST + s * 32 + hl * 8 + 16]);
      v16bf pa = cat8(lo, hi);
#pragma unroll
      for (int j = 0; j < 4; ++j) {
        v16bf vf = cat8(*reinterpret_cast<const v8bf*>(
            &ldsVt[(j * 16 + l16) * FST + s * 32 + hl * 8]),
            *reinterpret_cast<const v8bf*>(
            &ldsVt[(j * 16 + l16) * FST + s * 32 + hl * 8 + 16]));
        oacc[j] = __builtin_amdgcn_wmma_f32_16x16x32_bf16(
            false, pa, false, vf, (short)0, oacc[j], false, false);
      }
    }
  }
  // write output
#pragma unroll
  for (int j = 0; j < 4; ++j)
#pragma unroll
    for (int r = 0; r < 8; ++r) {
      const int q = q0 + wave * 16 + r + 8 * hl;
      O[((size_t)bb * S + q) * D + hh * HD + j * 16 + l16] = oacc[j][r] / lrow[r];
    }
}

// ---------------------------------------------------------------------------
// Router: softmax over 8 experts + top-2 renormalized combine weights
// ---------------------------------------------------------------------------
__global__ __launch_bounds__(128) void router_kernel(
    const float* __restrict__ tb, const float* __restrict__ Wr,
    float* __restrict__ cw, int T, int D) {
  const int row = blockIdx.x * 128 + threadIdx.x;
  if (row >= T) return;
  float lg[8];
#pragma unroll
  for (int e = 0; e < 8; ++e) lg[e] = 0.f;
  const float* tr = tb + (size_t)row * D;
  for (int d = 0; d < D; ++d) {
    const float tv = tr[d];
#pragma unroll
    for (int e = 0; e < 8; ++e) lg[e] += tv * Wr[d * 8 + e];
  }
  float mx = lg[0];
#pragma unroll
  for (int e = 1; e < 8; ++e) mx = fmaxf(mx, lg[e]);
  float p[8], se = 0.f;
#pragma unroll
  for (int e = 0; e < 8; ++e) { p[e] = __expf(lg[e] - mx); se += p[e]; }
#pragma unroll
  for (int e = 0; e < 8; ++e) p[e] /= se;
  int i1 = 0;
#pragma unroll
  for (int e = 1; e < 8; ++e) if (p[e] > p[i1]) i1 = e;
  int i2 = (i1 == 0) ? 1 : 0;
#pragma unroll
  for (int e = 0; e < 8; ++e) if (e != i1 && p[e] > p[i2]) i2 = e;
  const float s2 = p[i1] + p[i2];
#pragma unroll
  for (int e = 0; e < 8; ++e) cw[(size_t)row * 8 + e] = 0.f;
  cw[(size_t)row * 8 + i1] = p[i1] / s2;
  cw[(size_t)row * 8 + i2] = p[i2] / s2;
}

// ---------------------------------------------------------------------------
// act = silu(g) * u
// ---------------------------------------------------------------------------
__global__ __launch_bounds__(256) void silu_mul_kernel(
    const float* __restrict__ g, const float* __restrict__ u,
    float* __restrict__ out, size_t n) {
  const size_t i = (size_t)blockIdx.x * 256 + threadIdx.x;
  if (i < n) {
    const float gv = g[i];
    out[i] = (gv / (1.f + __expf(-gv))) * u[i];
  }
}

// ---------------------------------------------------------------------------
extern "C" void kernel_launch(void* const* d_in, const int* in_sizes, int n_in,
                              void* d_out, int out_size, void* d_ws, size_t ws_size,
                              hipStream_t stream) {
  (void)in_sizes; (void)n_in; (void)out_size; (void)ws_size;
  const float* x    = (const float*)d_in[0];
  const int*   mask = (const int*)  d_in[1];
  const float* Wq   = (const float*)d_in[2];
  const float* Wk   = (const float*)d_in[3];
  const float* Wv   = (const float*)d_in[4];
  const float* Wo   = (const float*)d_in[5];
  const float* Wr   = (const float*)d_in[6];
  const float* Wg   = (const float*)d_in[7];
  const float* Wu   = (const float*)d_in[8];
  const float* Wd   = (const float*)d_in[9];
  const float* g1   = (const float*)d_in[10];
  const float* g2   = (const float*)d_in[11];
  float* out = (float*)d_out;
  float* ws  = (float*)d_ws;

  const int Bn = 2, S = 1024, D = 1024, H = 2048, NH = 16, E = 8;
  const int T = Bn * S;                // 2048 tokens
  const size_t sz = (size_t)T * D;     // 2048*1024 floats

  float* A0 = ws;            // h, then t
  float* qb = ws + 1 * sz;
  float* kb = ws + 2 * sz;
  float* vb = ws + 3 * sz;
  float* ob = ws + 4 * sz;
  float* x1 = ws + 5 * sz;
  float* bufG = qb;          // [T,H] spans qb..kb (dead after attention)
  float* bufU = vb;          // [T,H] spans vb..ob
  float* bufA = x1;          // [T,H] spans x1..+2sz (x1 dead by then)
  float* cw   = ws + 7 * sz; // [T,E]

  // --- attention branch ---
  rmsnorm_kernel<<<T, 256, 0, stream>>>(x, g1, A0, D);
  dim3 gDD(D / 128, T / 128);
  gemm_bf16_wmma<0><<<gDD, 256, 0, stream>>>(A0, Wq, qb, nullptr, T, D, D, 0);
  gemm_bf16_wmma<0><<<gDD, 256, 0, stream>>>(A0, Wk, kb, nullptr, T, D, D, 0);
  gemm_bf16_wmma<0><<<gDD, 256, 0, stream>>>(A0, Wv, vb, nullptr, T, D, D, 0);
  dim3 gF(S / 64, Bn * NH);
  flash_attn_kernel<<<gF, 128, 0, stream>>>(qb, kb, vb, mask, ob);
  gemm_bf16_wmma<1><<<gDD, 256, 0, stream>>>(ob, Wo, x1, x, T, D, D, 0);

  // seed output with attention residual, then accumulate MoE into it
  hipMemcpyAsync(out, x1, sz * sizeof(float), hipMemcpyDeviceToDevice, stream);

  // --- MoE branch ---
  rmsnorm_kernel<<<T, 256, 0, stream>>>(x1, g2, A0, D);
  router_kernel<<<T / 128, 128, 0, stream>>>(A0, Wr, cw, T, D);
  dim3 gDH(H / 128, T / 128);
  for (int e = 0; e < E; ++e) {
    const float* Wge = Wg + (size_t)e * D * H;
    const float* Wue = Wu + (size_t)e * D * H;
    const float* Wde = Wd + (size_t)e * H * D;
    gemm_bf16_wmma<0><<<gDH, 256, 0, stream>>>(A0, Wge, bufG, nullptr, T, H, D, 0);
    gemm_bf16_wmma<0><<<gDH, 256, 0, stream>>>(A0, Wue, bufU, nullptr, T, H, D, 0);
    const size_t nTH = (size_t)T * H;
    silu_mul_kernel<<<(unsigned)((nTH + 255) / 256), 256, 0, stream>>>(bufG, bufU, bufA, nTH);
    gemm_bf16_wmma<2><<<gDD, 256, 0, stream>>>(bufA, Wde, out, cw + e, T, D, H, E);
  }
}
